// KNRM_25134148616775
// MI455X (gfx1250) — matrix operand order repaired and checked by reading, not compiled, not run on hardware
//
#include <hip/hip_runtime.h>

typedef __attribute__((ext_vector_type(16))) _Float16 v16h;
typedef __attribute__((ext_vector_type(8)))  _Float16 h8;
typedef __attribute__((ext_vector_type(4)))  _Float16 h4;
typedef __attribute__((ext_vector_type(8)))  float    v8f;

#define B_  512
#define W_  30
#define R_  512
#define D_  128
#define KN  21

// mu_i = (2i+1)/20 - 1, clamped to 1.0 for i=20 (sigma 0.001)
static constexpr float MU[KN] = {
    -0.95f,-0.85f,-0.75f,-0.65f,-0.55f,-0.45f,-0.35f,-0.25f,-0.15f,-0.05f,
     0.05f, 0.15f, 0.25f, 0.35f, 0.45f, 0.55f, 0.65f, 0.75f, 0.85f, 0.95f, 1.0f};
// -0.5 / sigma^2
static constexpr float NHI[KN] = {
    -50.f,-50.f,-50.f,-50.f,-50.f,-50.f,-50.f,-50.f,-50.f,-50.f,
    -50.f,-50.f,-50.f,-50.f,-50.f,-50.f,-50.f,-50.f,-50.f,-50.f, -500000.f};

// ---------------------------------------------------------------------------
// Kernel 1: gather embedding rows, L2-normalize, convert to f16.
// One wave per row; lane holds 4 consecutive floats (float4 load).
// ---------------------------------------------------------------------------
__global__ void __launch_bounds__(256)
gather_norm(const int* __restrict__ query, const int* __restrict__ doc,
            const float* __restrict__ emb,
            _Float16* __restrict__ qn, _Float16* __restrict__ dn)
{
    const int lane = threadIdx.x & 31;
    const int wv   = threadIdx.x >> 5;
    const int row  = blockIdx.x * 8 + wv;
    const int QROWS = B_ * W_;
    const int TOT   = QROWS + B_ * R_;
    if (row >= TOT) return;                 // uniform per wave

    int idx;
    _Float16* dst;
    if (row < QROWS) { idx = query[row];        dst = qn + (size_t)row * D_; }
    else             { int r = row - QROWS;
                       idx = doc[r];            dst = dn + (size_t)r   * D_; }

    const float4 x = ((const float4*)(emb + (size_t)idx * D_))[lane];
    float s = x.x*x.x + x.y*x.y + x.z*x.z + x.w*x.w;
    #pragma unroll
    for (int off = 16; off > 0; off >>= 1) s += __shfl_xor(s, off, 32);
    const float inv = 1.0f / fmaxf(sqrtf(s), 1e-12f);

    h4 o;
    o[0] = (_Float16)(x.x * inv);
    o[1] = (_Float16)(x.y * inv);
    o[2] = (_Float16)(x.z * inv);
    o[3] = (_Float16)(x.w * inv);
    ((h4*)dst)[lane] = o;                   // 8-byte store
}

// ---------------------------------------------------------------------------
// Kernel 2: per-batch WMMA GEMM q(30x128) x d^T(128x512) fused with the
// 21-kernel Gaussian pooling. One block (8 waves) per batch.
// Wave wv covers columns [wv*64, wv*64+64), both M-tiles (rows 0..31).
// ---------------------------------------------------------------------------
__global__ void __launch_bounds__(256)
knrm_match(const _Float16* __restrict__ qn, const _Float16* __restrict__ dn,
           float* __restrict__ ksum)
{
    const int b    = blockIdx.x;
    const int lane = threadIdx.x & 31;
    const int wv   = threadIdx.x >> 5;
    const int lo   = lane & 15;
    const int hi   = lane >> 4;

    const _Float16* qb = qn + (size_t)b * W_ * D_;
    const _Float16* db = dn + ((size_t)b * R_ + wv * 64) * D_;

    v8f acc[2][4];
    #pragma unroll
    for (int mt = 0; mt < 2; ++mt)
        #pragma unroll
        for (int nt = 0; nt < 4; ++nt) { v8f z = {}; acc[mt][nt] = z; }

    // A rows for the two M-tiles (tile1 rows 30,31 clamped; masked later)
    const int w0 = lo;
    int w1r = 16 + lo; if (w1r > W_ - 1) w1r = W_ - 1;
    const _Float16* aptr0 = qb + (size_t)w0  * D_ + hi * 8;
    const _Float16* aptr1 = qb + (size_t)w1r * D_ + hi * 8;

    #pragma unroll
    for (int kb = 0; kb < D_; kb += 32) {
        v16h a[2];
        {
            h8 a0l = *(const h8*)(aptr0 + kb);
            h8 a0h = *(const h8*)(aptr0 + kb + 16);
            h8 a1l = *(const h8*)(aptr1 + kb);
            h8 a1h = *(const h8*)(aptr1 + kb + 16);
            #pragma unroll
            for (int e = 0; e < 8; ++e) {
                a[0][e] = a0l[e]; a[0][e + 8] = a0h[e];
                a[1][e] = a1l[e]; a[1][e + 8] = a1h[e];
            }
        }
        #pragma unroll
        for (int nt = 0; nt < 4; ++nt) {
            const _Float16* bp = db + (size_t)(nt * 16 + lo) * D_ + kb + hi * 8;
            h8 bl = *(const h8*)bp;         // 2x16B: K table order per lane
            h8 bh = *(const h8*)(bp + 16);
            v16h bf;
            #pragma unroll
            for (int e = 0; e < 8; ++e) { bf[e] = bl[e]; bf[e + 8] = bh[e]; }
            #pragma unroll
            for (int mt = 0; mt < 2; ++mt)
                acc[mt][nt] = __builtin_amdgcn_wmma_f32_16x16x32_f16(
                    false, a[mt], false, bf, (short)0, acc[mt][nt], false, false);
        }
    }

    // Gaussian kernel pooling over the 64 M-elements held by this lane.
    float ks[KN];
    #pragma unroll
    for (int i = 0; i < KN; ++i) ks[i] = 0.0f;

    #pragma unroll
    for (int mt = 0; mt < 2; ++mt) {
        #pragma unroll
        for (int v = 0; v < 8; ++v) {
            const int w = mt * 16 + hi * 8 + v;   // global query-word index
            if (w < W_) {
                #pragma unroll
                for (int nt = 0; nt < 4; ++nt) {
                    const float x = acc[mt][nt][v];
                    #pragma unroll
                    for (int i = 0; i < KN; ++i) {
                        const float d = x - MU[i];
                        ks[i] += __expf(NHI[i] * d * d);
                    }
                }
            }
        }
    }

    // Wave reduction + global accumulation per (batch, kernel).
    #pragma unroll
    for (int i = 0; i < KN; ++i) {
        float v = ks[i];
        #pragma unroll
        for (int off = 16; off > 0; off >>= 1) v += __shfl_xor(v, off, 32);
        if (lane == 0) atomicAdd(&ksum[b * KN + i], v);
    }
}

// ---------------------------------------------------------------------------
// Kernel 3: log1p + MLP 21 -> 128 -> 64 -> 1. One block (128 threads) per batch.
// ---------------------------------------------------------------------------
__global__ void __launch_bounds__(128)
knrm_mlp(const float* __restrict__ ksum,
         const float* __restrict__ w1, const float* __restrict__ b1,
         const float* __restrict__ w2, const float* __restrict__ b2,
         const float* __restrict__ w3, const float* __restrict__ b3,
         float* __restrict__ out)
{
    const int b = blockIdx.x;
    const int t = threadIdx.x;
    __shared__ float k_s[KN];
    __shared__ float h1_s[128];
    __shared__ float h2_s[64];

    if (t < KN) k_s[t] = log1pf(ksum[b * KN + t]);
    __syncthreads();

    float h = b1[t];
    #pragma unroll
    for (int i = 0; i < KN; ++i) h += k_s[i] * w1[i * 128 + t];
    h1_s[t] = fmaxf(h, 0.0f);
    __syncthreads();

    if (t < 64) {
        float h2 = b2[t];
        for (int i = 0; i < 128; ++i) h2 += h1_s[i] * w2[i * 64 + t];
        h2_s[t] = fmaxf(h2, 0.0f);
    }
    __syncthreads();

    if (t == 0) {
        float o = b3[0];
        for (int j = 0; j < 64; ++j) o += h2_s[j] * w3[j];
        out[b] = o;
    }
}

// ---------------------------------------------------------------------------
extern "C" void kernel_launch(void* const* d_in, const int* in_sizes, int n_in,
                              void* d_out, int out_size, void* d_ws, size_t ws_size,
                              hipStream_t stream)
{
    const int*   query = (const int*)d_in[0];
    const int*   doc   = (const int*)d_in[1];
    const float* emb   = (const float*)d_in[2];
    const float* w1    = (const float*)d_in[3];
    const float* b1    = (const float*)d_in[4];
    const float* w2    = (const float*)d_in[5];
    const float* b2    = (const float*)d_in[6];
    const float* w3    = (const float*)d_in[7];
    const float* b3    = (const float*)d_in[8];
    float* out = (float*)d_out;

    char* ws = (char*)d_ws;
    _Float16* qn   = (_Float16*)(ws);                                   // 3.93 MB
    _Float16* dn   = (_Float16*)(ws + (size_t)4  * 1024 * 1024);        // 67.1 MB
    float*    ksum = (float*)   (ws + (size_t)72 * 1024 * 1024);        // 43 KB

    hipMemsetAsync(ksum, 0, (size_t)B_ * KN * sizeof(float), stream);

    const int TOT = B_ * W_ + B_ * R_;
    gather_norm<<<(TOT + 7) / 8, 256, 0, stream>>>(query, doc, emb, qn, dn);
    knrm_match<<<B_, 256, 0, stream>>>(qn, dn, ksum);
    knrm_mlp<<<B_, 128, 0, stream>>>(ksum, w1, b1, w2, b2, w3, b3, out);
}
